// MultiHeadedAttention_19713899889501
// MI455X (gfx1250) — matrix operand order repaired and checked by reading, not compiled, not run on hardware
//
#include <hip/hip_runtime.h>
#include <hip/hip_bf16.h>
#include <cstddef>
#include <cstdint>

// ---------------- problem constants ----------------
constexpr int B_    = 8;
constexpr int S_    = 2048;
constexpr int D_    = 512;
constexpr int H_    = 8;
constexpr int L_    = 5;
constexpr int DK_   = 64;
constexpr int SP_   = 2044;   // S - L + 1
constexpr int SPAD_ = 2048;   // padded rows (multiple of 64)

// ---------------- WMMA / TDM types ----------------
typedef __bf16 v16bf __attribute__((ext_vector_type(16)));
typedef float  v8f   __attribute__((ext_vector_type(8)));
typedef unsigned v4u __attribute__((ext_vector_type(4)));
typedef int      v8i __attribute__((ext_vector_type(8)));
typedef int      v4i __attribute__((ext_vector_type(4)));

union ABOp {
  v16bf v;
  unsigned u[8];
  uint4 q[2];
  unsigned short s[16];
};

#define DEV static __device__ __forceinline__

// f32 -> bf16 bits, round-to-nearest-even (no __bf16 arithmetic needed)
DEV unsigned short bf16u(float f) {
  unsigned u = __float_as_uint(f);
  return (unsigned short)((u + 0x7FFFu + ((u >> 16) & 1u)) >> 16);
}
DEV unsigned pack2(float a, float b) {
  return (unsigned)bf16u(a) | ((unsigned)bf16u(b) << 16);
}
// 16-bit A-matrix (16xK) K-offset for VGPR r, lane l  (ISA 7.12.2)
DEV int koffA(int r, int l) {
  return ((r < 4) ? (2 * r) : (16 + 2 * (r - 4))) + ((l & 16) ? 8 : 0);
}

// ---------------- TDM 2D tile load (ISA 8.3/8.4 descriptor) ----------------
// Copies a tile of (tile_y rows) x (tile_x contiguous 16-bit elems) from
// global (row pitch = stride_elems) into LDS at lds_addr, X-major.
DEV void tdm_load_2d(unsigned lds_addr, const unsigned short* gptr,
                     unsigned tile_x, unsigned tile_y,
                     unsigned stride_elems, unsigned dim0, unsigned dim1) {
  unsigned long long ga = (unsigned long long)(uintptr_t)gptr;
  v4u g0;
  g0[0] = 1u;                                   // count=1, user mode
  g0[1] = lds_addr;                             // LDS byte address
  g0[2] = (unsigned)ga;                         // global_addr[31:0]
  g0[3] = (unsigned)(ga >> 32) | (2u << 30);    // global_addr[56:32] | type=2
  v8i g1;
  g1[0] = (int)(1u << 16);                      // data_size = 2 bytes
  g1[1] = (int)((dim0 & 0xFFFFu) << 16);        // tensor_dim0[15:0]
  g1[2] = (int)(((dim0 >> 16) & 0xFFFFu) | ((dim1 & 0xFFFFu) << 16));
  g1[3] = (int)(((dim1 >> 16) & 0xFFFFu) | ((tile_x & 0xFFFFu) << 16));
  g1[4] = (int)(tile_y & 0xFFFFu);              // tile_dim1 (tile_dim2 = 0)
  g1[5] = (int)stride_elems;                    // tensor_dim0_stride[31:0]
  g1[6] = 0;
  g1[7] = 0;
  v4i z4 = {0, 0, 0, 0};
  v8i z8 = {0, 0, 0, 0, 0, 0, 0, 0};
  __builtin_amdgcn_tensor_load_to_lds(g0, g1, z4, z4, z8, 0);
}

// ---------------- kernel 1: convert W0/Wout to bf16 ----------------
__global__ void cvt_weights(const float* __restrict__ W0, const float* __restrict__ Wout,
                            unsigned short* __restrict__ w0b, unsigned short* __restrict__ wob) {
  int i = (blockIdx.x * blockDim.x + threadIdx.x) * 4;
  float4 a = *(const float4*)(W0 + i);
  float4 c = *(const float4*)(Wout + i);
  *(uint2*)(w0b + i) = make_uint2(pack2(a.x, a.y), pack2(a.z, a.w));
  *(uint2*)(wob + i) = make_uint2(pack2(c.x, c.y), pack2(c.z, c.w));
}

// ---------------- kernel 2: q local agg (window sum) -> bf16 ----------------
__global__ void qagg(const float* __restrict__ q, unsigned short* __restrict__ qb) {
  int idx = blockIdx.x * blockDim.x + threadIdx.x;   // one float4 each
  int d = (idx % (D_ / 4)) * 4;
  int t = (idx / (D_ / 4)) % SPAD_;
  int b = idx / ((D_ / 4) * SPAD_);
  float4 s = make_float4(0.f, 0.f, 0.f, 0.f);
  if (t < SP_) {
#pragma unroll
    for (int j = 0; j < L_; ++j) {
      float4 x = *(const float4*)(q + (size_t)(b * S_ + t + j) * D_ + d);
      s.x += x.x; s.y += x.y; s.z += x.z; s.w += x.w;
    }
  }
  *(uint2*)(qb + (size_t)(b * SPAD_ + t) * D_ + d) =
      make_uint2(pack2(s.x, s.y), pack2(s.z, s.w));
}

// ---------------- kernel 3: k local agg (softmax-weighted) -> bf16 ----------------
__global__ __launch_bounds__(256) void kagg(const float* __restrict__ k,
                                            unsigned short* __restrict__ kb) {
  int gw = blockIdx.x * 8 + (threadIdx.x >> 5);   // one wave per (b,t)
  int l  = threadIdx.x & 31;
  int b  = gw / SPAD_, t = gw % SPAD_;
  unsigned short* out = kb + (size_t)(b * SPAD_ + t) * D_ + l * 16;
  if (t >= SP_) {
    uint4 z = make_uint4(0u, 0u, 0u, 0u);
    *(uint4*)out = z; *(uint4*)(out + 8) = z;
    return;
  }
  const float* base = k + (size_t)(b * S_ + t) * D_ + l * 16;  // 16 d's per lane
  float xc[16];
#pragma unroll
  for (int i = 0; i < 4; ++i) {
    float4 v = *(const float4*)(base + (L_ - 1) * D_ + i * 4);
    xc[i * 4 + 0] = v.x; xc[i * 4 + 1] = v.y; xc[i * 4 + 2] = v.z; xc[i * 4 + 3] = v.w;
  }
  float dots[L_];
#pragma unroll
  for (int j = 0; j < L_; ++j) {
    float p = 0.f;
#pragma unroll
    for (int i = 0; i < 4; ++i) {
      float4 v = *(const float4*)(base + j * D_ + i * 4);
      p += xc[i * 4 + 0] * v.x + xc[i * 4 + 1] * v.y + xc[i * 4 + 2] * v.z + xc[i * 4 + 3] * v.w;
    }
    dots[j] = p;
  }
  // batched 32-lane reductions (5 independent bpermute chains per round)
#pragma unroll
  for (int m = 1; m < 32; m <<= 1) {
    float t5[L_];
#pragma unroll
    for (int j = 0; j < L_; ++j) t5[j] = __shfl_xor(dots[j], m, 32);
#pragma unroll
    for (int j = 0; j < L_; ++j) dots[j] += t5[j];
  }
  const float inv = 0.04419417382415922f;  // 1/sqrt(512)
  float mx = -1e30f;
#pragma unroll
  for (int j = 0; j < L_; ++j) mx = fmaxf(mx, dots[j] * inv);
  float wj[L_], wsum = 0.f;
#pragma unroll
  for (int j = 0; j < L_; ++j) { wj[j] = __expf(dots[j] * inv - mx); wsum += wj[j]; }
  float r = 1.f / wsum;
  float acc[16];
#pragma unroll
  for (int i = 0; i < 16; ++i) acc[i] = 0.f;
#pragma unroll
  for (int j = 0; j < L_; ++j) {
    float wv = wj[j] * r;
#pragma unroll
    for (int i = 0; i < 4; ++i) {
      float4 v = *(const float4*)(base + j * D_ + i * 4);
      acc[i * 4 + 0] += wv * v.x; acc[i * 4 + 1] += wv * v.y;
      acc[i * 4 + 2] += wv * v.z; acc[i * 4 + 3] += wv * v.w;
    }
  }
  *(uint4*)out = make_uint4(pack2(acc[0], acc[1]), pack2(acc[2], acc[3]),
                            pack2(acc[4], acc[5]), pack2(acc[6], acc[7]));
  *(uint4*)(out + 8) = make_uint4(pack2(acc[8], acc[9]), pack2(acc[10], acc[11]),
                                  pack2(acc[12], acc[13]), pack2(acc[14], acc[15]));
}

// ---------------- kernel 3b: transpose k_bf -> kT [b][h][dk][t] ----------------
__global__ __launch_bounds__(256) void ktr(const unsigned short* __restrict__ kbf,
                                           unsigned short* __restrict__ kT) {
  __shared__ unsigned short tile[64][65];   // +1 pad against bank conflicts
  int blk = blockIdx.x;                     // B*H*(SPAD/64)
  int tb = blk & 31;
  int h  = (blk >> 5) & 7;
  int b  = blk >> 8;
  int t0 = tb * 64;
  {
    int row = threadIdx.x >> 2;             // t offset 0..63
    int c0  = (threadIdx.x & 3) * 16;       // dk offset
    const unsigned short* src = kbf + (size_t)(b * SPAD_ + t0 + row) * D_ + h * DK_ + c0;
    union { uint4 q[2]; unsigned short s[16]; } tmp;
    tmp.q[0] = ((const uint4*)src)[0];
    tmp.q[1] = ((const uint4*)src)[1];
#pragma unroll
    for (int i = 0; i < 16; ++i) tile[row][c0 + i] = tmp.s[i];
  }
  __syncthreads();
  {
    int dk = threadIdx.x >> 2;              // 0..63
    int tc = (threadIdx.x & 3) * 16;        // t offset
    union { uint4 q[2]; unsigned short s[16]; } o;
#pragma unroll
    for (int i = 0; i < 16; ++i) o.s[i] = tile[tc + i][dk];
    unsigned short* dst = kT + (size_t)((b * H_ + h) * DK_ + dk) * SPAD_ + t0 + tc;
    ((uint4*)dst)[0] = o.q[0];
    ((uint4*)dst)[1] = o.q[1];
  }
}

// ---------------- kernel 4/6: WMMA GEMM  out = in[t+rowOff,:] @ Wb + bias ----------------
template <int MODE>
__global__ __launch_bounds__(128) void gemm512(const float* __restrict__ in,
                                               const unsigned short* __restrict__ Wb,
                                               const float* __restrict__ bias,
                                               void* __restrict__ outp, int rowOff) {
  __shared__ unsigned short sA[4][16 * 32];
  const int tid = threadIdx.x, w = tid >> 5, l = tid & 31;
  const int b = blockIdx.x >> 5, mt = blockIdx.x & 31;
  const int m0 = mt * 64 + w * 16;
  const int nn = l & 15, hi = (l >> 4) & 1;
  const int lr = l >> 1, ksub = (l & 1) * 16;
  const int t_ld = m0 + lr;
  const float* arow = in + (size_t)(b * S_ + t_ld + rowOff) * D_;

  for (int nb = 0; nb < D_; nb += 64) {
    v8f acc[4];
#pragma unroll
    for (int c = 0; c < 4; ++c)
#pragma unroll
      for (int r = 0; r < 8; ++r) acc[c][r] = 0.f;

    for (int kb = 0; kb < D_; kb += 32) {
      unsigned au[8];
      if (t_ld < SP_) {
#pragma unroll
        for (int i = 0; i < 4; ++i) {
          float4 v = *(const float4*)(arow + kb + ksub + i * 4);
          au[i * 2]     = pack2(v.x, v.y);
          au[i * 2 + 1] = pack2(v.z, v.w);
        }
      } else {
#pragma unroll
        for (int i = 0; i < 8; ++i) au[i] = 0u;
      }
      uint4* ad = (uint4*)&sA[w][lr * 32 + ksub];
      ad[0] = make_uint4(au[0], au[1], au[2], au[3]);
      ad[1] = make_uint4(au[4], au[5], au[6], au[7]);

      ABOp aop;
#pragma unroll
      for (int r = 0; r < 8; ++r)
        aop.u[r] = *(const unsigned*)&sA[w][nn * 32 + koffA(r, l)];

      const unsigned short* wrow = Wb + (size_t)kb * D_ + nb + nn;
#pragma unroll
      for (int c = 0; c < 4; ++c) {
        ABOp bop;
#pragma unroll
        for (int r = 0; r < 8; ++r) {
          bop.s[2 * r]     = wrow[(size_t)koffA(r, l) * D_ + c * 16];
          bop.s[2 * r + 1] = wrow[(size_t)(koffA(r, l) + 1) * D_ + c * 16];
        }
        acc[c] = __builtin_amdgcn_wmma_f32_16x16x32_bf16(
            false, aop.v, false, bop.v, (short)0, acc[c], false, false);
      }
    }
#pragma unroll
    for (int c = 0; c < 4; ++c) {
      int n = nb + c * 16 + nn;
      float bv = bias[n];
#pragma unroll
      for (int r = 0; r < 8; ++r) {
        int t = m0 + r + 8 * hi;
        float val = acc[c][r] + bv;
        if (MODE == 0) {
          ((unsigned short*)outp)[(size_t)(b * SPAD_ + t) * D_ + n] = bf16u(val);
        } else {
          if (t < SP_) ((float*)outp)[(size_t)(b * SP_ + t) * D_ + n] = val;
        }
      }
    }
  }
}

// ---------------- kernel 5: flash attention (bf16 WMMA + TDM staging) ----------------
__global__ __launch_bounds__(128) void flash_attn(const unsigned short* __restrict__ qb,
                                                  const unsigned short* __restrict__ kT,
                                                  const unsigned short* __restrict__ vbf,
                                                  float* __restrict__ x) {
  __shared__ unsigned short sKT[2][DK_ * 32];  // K^T tile: [dk(64)][key(32)], double-buffered
  __shared__ unsigned short sV[2][32 * DK_];   // V tile:   [key(32)][dk(64)], double-buffered
  __shared__ unsigned short sQ[4][16 * DK_];   // per-wave Q tile [m][dk]
  __shared__ unsigned short sP[4][16 * 32];    // per-wave P tile [m][key]

  const int tid = threadIdx.x, w = tid >> 5, l = tid & 31;
  const int bh = blockIdx.x >> 5, mt = blockIdx.x & 31;
  const int b = bh >> 3, h = bh & 7;
  const int m0 = mt * 64 + w * 16;
  const int nn = l & 15, hi = (l >> 4) & 1;

  const unsigned short* kTbase = kT + (size_t)((b * H_ + h) * DK_) * SPAD_;  // [dk][t]
  const unsigned short* vbase  = vbf + (size_t)(b * SPAD_) * D_ + h * DK_;   // [t][dk]

  {  // stage this wave's 16x64 Q tile
    int row = l >> 1, part = l & 1;
    const unsigned short* src = qb + (size_t)(b * SPAD_ + m0 + row) * D_ + h * DK_ + part * 32;
    uint4* dst = (uint4*)&sQ[w][row * DK_ + part * 32];
#pragma unroll
    for (int i = 0; i < 4; ++i) dst[i] = ((const uint4*)src)[i];
  }
  ABOp qop[2];
#pragma unroll
  for (int o = 0; o < 2; ++o)
#pragma unroll
    for (int r = 0; r < 8; ++r)
      qop[o].u[r] = *(const unsigned*)&sQ[w][nn * DK_ + 32 * o + koffA(r, l)];

  v8f O[4];
  float mi[8], li[8];
#pragma unroll
  for (int c = 0; c < 4; ++c)
#pragma unroll
    for (int r = 0; r < 8; ++r) O[c][r] = 0.f;
#pragma unroll
  for (int r = 0; r < 8; ++r) { mi[r] = -1e30f; li[r] = 0.f; }

  constexpr int NT = SPAD_ / 32;  // 64 key tiles
  if (w == 0) {  // prologue: DMA tile 0 (wave-level op; TDM ignores EXEC)
    tdm_load_2d((unsigned)(uintptr_t)&sKT[0][0], kTbase, 32, DK_, SPAD_, SPAD_, DK_);
    tdm_load_2d((unsigned)(uintptr_t)&sV[0][0], vbase, DK_, 32, D_, D_, SPAD_);
  }

  for (int i = 0; i < NT; ++i) {
    const int kb = i * 32, cur = i & 1;
    if (w == 0) {
      if (i + 1 < NT) {
        tdm_load_2d((unsigned)(uintptr_t)&sKT[cur ^ 1][0], kTbase + (kb + 32),
                    32, DK_, SPAD_, SPAD_, DK_);
        tdm_load_2d((unsigned)(uintptr_t)&sV[cur ^ 1][0], vbase + (size_t)(kb + 32) * D_,
                    DK_, 32, D_, D_, SPAD_);
        __builtin_amdgcn_s_wait_tensorcnt(2);  // tile i complete; tile i+1 in flight
      } else {
        __builtin_amdgcn_s_wait_tensorcnt(0);
      }
    }
    __syncthreads();  // tile i visible to all waves

    // S = Q @ K^T  (16 queries x 32 keys), contraction over dk=64
    v8f Sacc[2];
#pragma unroll
    for (int kc = 0; kc < 2; ++kc)
#pragma unroll
      for (int r = 0; r < 8; ++r) Sacc[kc][r] = 0.f;
#pragma unroll
    for (int kc = 0; kc < 2; ++kc) {
#pragma unroll
      for (int cb = 0; cb < 2; ++cb) {
        ABOp bop;
#pragma unroll
        for (int r = 0; r < 8; ++r) {
          bop.s[2 * r]     = sKT[cur][(cb * 32 + koffA(r, l)) * 32 + kc * 16 + nn];
          bop.s[2 * r + 1] = sKT[cur][(cb * 32 + koffA(r, l) + 1) * 32 + kc * 16 + nn];
        }
        Sacc[kc] = __builtin_amdgcn_wmma_f32_16x16x32_bf16(
            false, qop[cb].v, false, bop.v, (short)0, Sacc[kc], false, false);
      }
    }

    // online softmax; reductions batched across the 8 rows for MLP
    float s0[8], s1[8], rm[8];
#pragma unroll
    for (int r = 0; r < 8; ++r) {
      s0[r] = Sacc[0][r] * 0.125f;  // 1/sqrt(DK)
      s1[r] = Sacc[1][r] * 0.125f;
      if (kb + nn >= SP_)      s0[r] = -1e30f;
      if (kb + 16 + nn >= SP_) s1[r] = -1e30f;
      rm[r] = fmaxf(s0[r], s1[r]);
    }
#pragma unroll
    for (int m = 1; m < 16; m <<= 1) {
      float t8[8];
#pragma unroll
      for (int r = 0; r < 8; ++r) t8[r] = __shfl_xor(rm[r], m, 32);
#pragma unroll
      for (int r = 0; r < 8; ++r) rm[r] = fmaxf(rm[r], t8[r]);
    }
    float corr[8], ps[8];
#pragma unroll
    for (int r = 0; r < 8; ++r) {
      float mnew = fmaxf(mi[r], rm[r]);
      corr[r] = __expf(mi[r] - mnew);
      float p0 = __expf(s0[r] - mnew);
      float p1 = __expf(s1[r] - mnew);
      mi[r] = mnew;
      ps[r] = p0 + p1;
      sP[w][(r + 8 * hi) * 32 + nn]      = bf16u(p0);
      sP[w][(r + 8 * hi) * 32 + 16 + nn] = bf16u(p1);
    }
#pragma unroll
    for (int m = 1; m < 16; m <<= 1) {
      float t8[8];
#pragma unroll
      for (int r = 0; r < 8; ++r) t8[r] = __shfl_xor(ps[r], m, 32);
#pragma unroll
      for (int r = 0; r < 8; ++r) ps[r] += t8[r];
    }
#pragma unroll
    for (int r = 0; r < 8; ++r) {
      li[r] = li[r] * corr[r] + ps[r];
#pragma unroll
      for (int c = 0; c < 4; ++c) O[c][r] *= corr[r];
    }

    // O += P @ V  (A = P 16x32 via LDS round-trip, B = V rows direct)
    ABOp pop;
#pragma unroll
    for (int r = 0; r < 8; ++r)
      pop.u[r] = *(const unsigned*)&sP[w][nn * 32 + koffA(r, l)];
#pragma unroll
    for (int c = 0; c < 4; ++c) {
      ABOp vop;
#pragma unroll
      for (int r = 0; r < 8; ++r) {
        vop.s[2 * r]     = sV[cur][koffA(r, l) * DK_ + c * 16 + nn];
        vop.s[2 * r + 1] = sV[cur][(koffA(r, l) + 1) * DK_ + c * 16 + nn];
      }
      O[c] = __builtin_amdgcn_wmma_f32_16x16x32_bf16(
          false, pop.v, false, vop.v, (short)0, O[c], false, false);
    }
    __syncthreads();  // all waves done with buffer `cur` before it is re-filled
  }

#pragma unroll
  for (int r = 0; r < 8; ++r) {
    int t = m0 + r + 8 * hi;
    if (t >= SP_) continue;
    float invl = 1.0f / li[r];
#pragma unroll
    for (int c = 0; c < 4; ++c)
      x[(size_t)(b * SPAD_ + t) * D_ + h * DK_ + c * 16 + nn] = O[c][r] * invl;
  }
}

// ---------------- launcher ----------------
extern "C" void kernel_launch(void* const* d_in, const int* in_sizes, int n_in,
                              void* d_out, int out_size, void* d_ws, size_t ws_size,
                              hipStream_t stream) {
  (void)in_sizes; (void)n_in; (void)out_size; (void)ws_size;
  const float* query = (const float*)d_in[0];
  const float* key_  = (const float*)d_in[1];
  const float* value = (const float*)d_in[2];
  const float* W0    = (const float*)d_in[3];
  const float* b0    = (const float*)d_in[4];
  const float* Wout  = (const float*)d_in[5];
  const float* bout  = (const float*)d_in[6];

  const size_t NBF = (size_t)B_ * SPAD_ * D_;  // padded elements per tensor
  unsigned short* qbf = (unsigned short*)d_ws;
  unsigned short* kbf = qbf + NBF;
  unsigned short* vbf = kbf + NBF;
  unsigned short* kTt = vbf + NBF;             // transposed K: [b][h][dk][t]
  unsigned short* w0b = kTt + NBF;
  unsigned short* wob = w0b + (size_t)D_ * D_;
  float* x = (float*)(wob + (size_t)D_ * D_);  // byte offset is 16B-aligned

  cvt_weights<<<dim3((D_ * D_ / 4) / 256), dim3(256), 0, stream>>>(W0, Wout, w0b, wob);
  qagg<<<dim3((B_ * SPAD_ * (D_ / 4)) / 256), dim3(256), 0, stream>>>(query, qbf);
  kagg<<<dim3((B_ * SPAD_) / 8), dim3(256), 0, stream>>>(key_, kbf);
  ktr<<<dim3(B_ * H_ * (SPAD_ / 64)), dim3(256), 0, stream>>>(kbf, kTt);
  gemm512<0><<<dim3(B_ * 32), dim3(128), 0, stream>>>(value, w0b, b0, (void*)vbf, L_ - 1);
  flash_attn<<<dim3(B_ * H_ * 32), dim3(128), 0, stream>>>(qbf, kTt, vbf, x);
  gemm512<1><<<dim3(B_ * 32), dim3(128), 0, stream>>>(x, wob, bout, d_out, 0);
}
